// Decoder_75411035783388
// MI455X (gfx1250) — matrix operand order repaired and checked by reading, not compiled, not run on hardware
//
#include <hip/hip_runtime.h>
#include <hip/hip_bf16.h>
#include <math.h>

// ---------------------------------------------------------------------------
// MI455X (gfx1250, wave32) fused ProteinMPNN-decoder.
//   N=8192 nodes, K=48 neighbors, C=128, L=3 layers.
//   All GEMMs via v_wmma_f32_16x16x32_f16 (f16 in, f32 accumulate).
//   One workgroup (256 thr = 8 waves) per node; all 3 layers fused (no
//   cross-node dependence after the initial neighbor gather).
//   Weights pre-repacked (fp32 -> f16) into WMMA B-fragment register order
//   in d_ws (~1.31 MB, L2 resident).
// ---------------------------------------------------------------------------

typedef _Float16 v4h  __attribute__((ext_vector_type(4)));
typedef _Float16 v8h  __attribute__((ext_vector_type(8)));
typedef _Float16 v16h __attribute__((ext_vector_type(16)));
typedef float    v8f  __attribute__((ext_vector_type(8)));

#define KNBR 48
#define CCH  128
#define LDA  520   // 48 x 520 f16 : message A staging (512 cols + 8 pad)
#define LDM  136   // 48 x 136 f16 : message intermediate staging
#define LDH  136   // 16 x 136 f16 : dense1 A staging (row 0 live)
#define LDD  520   // 16 x 520 f16 : dense2 A staging (row 0 live)

// fragment table (per layer, fragment = 1024 B = 32 lanes x 32 B):
//   W1 : frags [  0,128)  ksteps=16, ntiles=8   (512x128)
//   W2 : frags [128,160)  ksteps= 4, ntiles=8   (128x128)
//   W3 : frags [160,192)  ksteps= 4, ntiles=8   (128x128)
//   D1 : frags [192,320)  ksteps= 4, ntiles=32  (128x512)
//   D2 : frags [320,448)  ksteps=16, ntiles=8   (512x128)
#define FRAGS_PER_LAYER 448

// --------------------------- weight repack ---------------------------------
// B-fragment layout for v_wmma_f32_16x16x32_f16 (B is 32x16):
//   lane l: col n = ntile*16 + (l&15);  halves j=0..7 hold K = kbase + (l>>4)*8 + j, halves 8..15 hold K +16
__global__ __launch_bounds__(32) void repack_kernel(
    const float* __restrict__ msg_w1, const float* __restrict__ msg_w2,
    const float* __restrict__ msg_w3, const float* __restrict__ dense_w1,
    const float* __restrict__ dense_w2, _Float16* __restrict__ ws)
{
    const int b = blockIdx.x;
    const int layer = b / FRAGS_PER_LAYER;
    const int f = b % FRAGS_PER_LAYER;
    const float* src; int Nc, kstep, ntile;
    if (f < 128)      { int g = f;       src = msg_w1   + layer*512*128; Nc = 128; kstep = g >> 3; ntile = g & 7; }
    else if (f < 160) { int g = f - 128; src = msg_w2   + layer*128*128; Nc = 128; kstep = g >> 3; ntile = g & 7; }
    else if (f < 192) { int g = f - 160; src = msg_w3   + layer*128*128; Nc = 128; kstep = g >> 3; ntile = g & 7; }
    else if (f < 320) { int g = f - 192; src = dense_w1 + layer*128*512; Nc = 512; kstep = g >> 5; ntile = g & 31; }
    else              { int g = f - 320; src = dense_w2 + layer*512*128; Nc = 128; kstep = g >> 3; ntile = g & 7; }
    const int lane = threadIdx.x;
    const int col  = ntile*16 + (lane & 15);
    const int kb   = kstep*32 + (lane >> 4)*8;
    _Float16* dst = ws + (size_t)b*512 + lane*16;   // 512 halves/frag, 16 halves/lane
    v8h lo, hi;
    #pragma unroll
    for (int j = 0; j < 8; ++j) {
        lo[j] = (_Float16)src[(size_t)(kb + j)     * Nc + col];
        hi[j] = (_Float16)src[(size_t)(kb + 16 + j) * Nc + col];
    }
    *(v8h*)dst       = lo;
    *(v8h*)(dst + 8) = hi;
}

// --------------------------- device helpers --------------------------------
__device__ __forceinline__ v8f wmma16(v16h a, v16h b, v8f c) {
    return __builtin_amdgcn_wmma_f32_16x16x32_f16(false, a, false, b, (short)0, c, false, false);
}

// A fragment (16x32 f16) from row-major LDS: lane l row = mtile*16+(l&15);
// halves 0..7 = cols kbase+(l>>4)*8+{0..7}, halves 8..15 = cols +16.
__device__ __forceinline__ v16h ldsA(const _Float16* base, int ld, int mtile,
                                     int kbase, int lane) {
    const int row = mtile*16 + (lane & 15);
    const _Float16* p = base + row*ld + kbase + (lane >> 4)*8;
    union { v16h v; v8h h[2]; } u;
    u.h[0] = *(const v8h*)p;
    u.h[1] = *(const v8h*)(p + 16);
    return u.v;
}

// B fragment straight from repacked global weights: lane reads its 32 bytes.
__device__ __forceinline__ v16h ldB(const _Float16* ws, int frag, int lane) {
    const _Float16* p = ws + (size_t)frag*512 + lane*16;
    union { v16h v; v8h h[2]; } u;
    u.h[0] = *(const v8h*)p;
    u.h[1] = *(const v8h*)(p + 8);
    return u.v;
}

__device__ __forceinline__ float gelu_exact(float x) {
    return 0.5f * x * (1.0f + erff(x * 0.70710678118654752f));
}

// ------------------------------ main kernel --------------------------------
__global__ __launch_bounds__(256) void decoder_kernel(
    const float* __restrict__ nodef, const float* __restrict__ edgef,
    const int*   __restrict__ nidx,  const float* __restrict__ maskp,
    const float* __restrict__ msg_b1, const float* __restrict__ msg_b2,
    const float* __restrict__ msg_b3,
    const float* __restrict__ n1_scale, const float* __restrict__ n1_bias,
    const float* __restrict__ dense_b1, const float* __restrict__ dense_b2,
    const float* __restrict__ n2_scale, const float* __restrict__ n2_bias,
    const _Float16* __restrict__ wsf, float* __restrict__ out)
{
    extern __shared__ char smem_raw[];
    _Float16* sA = (_Float16*)smem_raw;      // 48*520 halves
    _Float16* sM = sA + 48*LDA;              // 48*136
    _Float16* sH = sM + 48*LDM;              // 16*136
    _Float16* sD = sH + 16*LDH;              // 16*520
    float* hvec   = (float*)(sD + 16*LDD);   // 128
    float* hnbuf  = hvec  + 128;             // 128
    float* h2buf  = hnbuf + 128;             // 128
    float* aggbuf = h2buf + 128;             // 256
    float* red    = aggbuf + 256;            // 128
    float* red2   = red + 128;               // 128

    const int tid  = threadIdx.x;
    const int wave = tid >> 5;
    const int lane = tid & 31;
    const int n    = blockIdx.x;

    // ---- stage constant A columns (f32 -> f16):
    //      cols 128..255 = edge_features[n,k,:], cols 384..511 = nodes[idx[n,k],:]
    //      cols 256..383 are an all-zero block: GEMM1 simply skips those k-steps.
    for (int e = tid; e < KNBR*32; e += 256) {
        const int r = e >> 5, c4 = e & 31;
        const float4 fe = ((const float4*)(edgef + ((size_t)n*KNBR + r)*CCH))[c4];
        v4h ev = { (_Float16)fe.x, (_Float16)fe.y, (_Float16)fe.z, (_Float16)fe.w };
        *(v4h*)&sA[r*LDA + 128 + c4*4] = ev;
        const int nb = nidx[n*KNBR + r];
        const float4 fg = ((const float4*)(nodef + (size_t)nb*CCH))[c4];
        v4h gv = { (_Float16)fg.x, (_Float16)fg.y, (_Float16)fg.z, (_Float16)fg.w };
        *(v4h*)&sA[r*LDA + 384 + c4*4] = gv;
    }
    // zero dense A-staging (rows 1..15 must be 0; row 0 rewritten each layer)
    for (int e = tid; e < 16*LDH; e += 256) sH[e] = (_Float16)0.f;
    for (int e = tid; e < 16*LDD; e += 256) sD[e] = (_Float16)0.f;
    if (tid < 128) hvec[tid] = nodef[(size_t)n*CCH + tid];
    const float mk = maskp[n];
    __syncthreads();

    for (int layer = 0; layer < 3; ++layer) {
        const int fb = layer * FRAGS_PER_LAYER;

        // broadcast current h into A cols 0..127 (all 48 rows)
        for (int e = tid; e < KNBR*32; e += 256) {
            const int r = e >> 5, c4 = e & 31;
            v4h hv = { (_Float16)hvec[c4*4+0], (_Float16)hvec[c4*4+1],
                       (_Float16)hvec[c4*4+2], (_Float16)hvec[c4*4+3] };
            *(v4h*)&sA[r*LDA + c4*4] = hv;
        }
        __syncthreads();

        // ======== message GEMM1: (48 x 512) x (512 x 128), zero block skipped
        {
            v8f a0 = {}, a1 = {}, a2 = {};
            #pragma unroll
            for (int ks = 0; ks < 16; ++ks) {
                if (ks >= 8 && ks < 12) continue;          // zero columns 256..383
                const v16h b = ldB(wsf, fb + ks*8 + wave, lane);
                a0 = wmma16(ldsA(sA, LDA, 0, ks*32, lane), b, a0);
                a1 = wmma16(ldsA(sA, LDA, 1, ks*32, lane), b, a1);
                a2 = wmma16(ldsA(sA, LDA, 2, ks*32, lane), b, a2);
            }
            const int nc = wave*16 + (lane & 15);
            const float b1 = msg_b1[layer*CCH + nc];
            const int s = lane >> 4;
            #pragma unroll
            for (int j = 0; j < 8; ++j) {
                sM[( 0 + j + 8*s)*LDM + nc] = (_Float16)gelu_exact(a0[j] + b1);
                sM[(16 + j + 8*s)*LDM + nc] = (_Float16)gelu_exact(a1[j] + b1);
                sM[(32 + j + 8*s)*LDM + nc] = (_Float16)gelu_exact(a2[j] + b1);
            }
        }
        __syncthreads();

        // ======== message GEMM2: (48 x 128) x (128 x 128)
        {
            v8f a0 = {}, a1 = {}, a2 = {};
            #pragma unroll
            for (int ks = 0; ks < 4; ++ks) {
                const v16h b = ldB(wsf, fb + 128 + ks*8 + wave, lane);
                a0 = wmma16(ldsA(sM, LDM, 0, ks*32, lane), b, a0);
                a1 = wmma16(ldsA(sM, LDM, 1, ks*32, lane), b, a1);
                a2 = wmma16(ldsA(sM, LDM, 2, ks*32, lane), b, a2);
            }
            __syncthreads();                     // everyone done reading sM
            const int nc = wave*16 + (lane & 15);
            const float b2 = msg_b2[layer*CCH + nc];
            const int s = lane >> 4;
            #pragma unroll
            for (int j = 0; j < 8; ++j) {
                sM[( 0 + j + 8*s)*LDM + nc] = (_Float16)gelu_exact(a0[j] + b2);
                sM[(16 + j + 8*s)*LDM + nc] = (_Float16)gelu_exact(a1[j] + b2);
                sM[(32 + j + 8*s)*LDM + nc] = (_Float16)gelu_exact(a2[j] + b2);
            }
        }
        __syncthreads();

        // ======== message GEMM3 + row-sum aggregation over the 48 edges
        {
            v8f a0 = {}, a1 = {}, a2 = {};
            #pragma unroll
            for (int ks = 0; ks < 4; ++ks) {
                const v16h b = ldB(wsf, fb + 160 + ks*8 + wave, lane);
                a0 = wmma16(ldsA(sM, LDM, 0, ks*32, lane), b, a0);
                a1 = wmma16(ldsA(sM, LDM, 1, ks*32, lane), b, a1);
                a2 = wmma16(ldsA(sM, LDM, 2, ks*32, lane), b, a2);
            }
            float part = 0.f;
            #pragma unroll
            for (int j = 0; j < 8; ++j) part += a0[j] + a1[j] + a2[j];
            aggbuf[(lane >> 4)*128 + wave*16 + (lane & 15)] = part;
        }
        __syncthreads();
        if (tid < 128) {
            const float msum = aggbuf[tid] + aggbuf[128 + tid]
                             + 48.0f * msg_b3[layer*CCH + tid];   // bias x K edges
            hvec[tid] += msum * (1.0f / 30.0f);
        }
        __syncthreads();

        // ======== LayerNorm 1
        float hv = (tid < 128) ? hvec[tid] : 0.f;
        if (tid < 128) { red[tid] = hv; red2[tid] = hv*hv; }
        __syncthreads();
        for (int sr = 64; sr > 0; sr >>= 1) {
            if (tid < sr) { red[tid] += red[tid+sr]; red2[tid] += red2[tid+sr]; }
            __syncthreads();
        }
        {
            const float mu  = red[0] * (1.0f/128.0f);
            const float var = red2[0] * (1.0f/128.0f) - mu*mu;
            const float rs  = rsqrtf(var + 1e-5f);
            if (tid < 128) {
                const float hn = (hv - mu) * rs * n1_scale[layer*CCH + tid]
                               + n1_bias[layer*CCH + tid];
                hnbuf[tid] = hn;
                sH[tid] = (_Float16)hn;          // row 0 of dense1 A
            }
        }
        __syncthreads();

        // ======== dense1 via WMMA: (16 x 128) x (128 x 512), only row 0 live
        #pragma unroll
        for (int nt4 = 0; nt4 < 4; ++nt4) {
            const int ntile = wave + nt4*8;
            v8f acc = {};
            #pragma unroll
            for (int ks = 0; ks < 4; ++ks)
                acc = wmma16(ldsA(sH, LDH, 0, ks*32, lane),
                             ldB(wsf, fb + 192 + ks*32 + ntile, lane), acc);
            if (lane < 16) {
                const int col = ntile*16 + lane;
                sD[col] = (_Float16)gelu_exact(acc[0] + dense_b1[layer*512 + col]);
            }
        }
        __syncthreads();

        // ======== dense2 via WMMA: (16 x 512) x (512 x 128), only row 0 live
        {
            v8f acc = {};
            #pragma unroll
            for (int ks = 0; ks < 16; ++ks)
                acc = wmma16(ldsA(sD, LDD, 0, ks*32, lane),
                             ldB(wsf, fb + 320 + ks*8 + wave, lane), acc);
            if (lane < 16) {
                const int col = wave*16 + lane;
                h2buf[col] = hnbuf[col] + acc[0] + dense_b2[layer*CCH + col];
            }
        }
        __syncthreads();

        // ======== LayerNorm 2 + mask
        float h2 = (tid < 128) ? h2buf[tid] : 0.f;
        if (tid < 128) { red[tid] = h2; red2[tid] = h2*h2; }
        __syncthreads();
        for (int sr = 64; sr > 0; sr >>= 1) {
            if (tid < sr) { red[tid] += red[tid+sr]; red2[tid] += red2[tid+sr]; }
            __syncthreads();
        }
        {
            const float mu  = red[0] * (1.0f/128.0f);
            const float var = red2[0] * (1.0f/128.0f) - mu*mu;
            const float rs  = rsqrtf(var + 1e-5f);
            if (tid < 128)
                hvec[tid] = mk * ((h2 - mu) * rs * n2_scale[layer*CCH + tid]
                                  + n2_bias[layer*CCH + tid]);
        }
        __syncthreads();
    }

    if (tid < 128) out[(size_t)n*CCH + tid] = hvec[tid];
}

// ------------------------------- launcher ----------------------------------
extern "C" void kernel_launch(void* const* d_in, const int* in_sizes, int n_in,
                              void* d_out, int out_size, void* d_ws, size_t ws_size,
                              hipStream_t stream) {
    (void)n_in; (void)out_size; (void)ws_size;
    const float* nodef    = (const float*)d_in[0];
    const float* edgef    = (const float*)d_in[1];
    const int*   nidx     = (const int*)  d_in[2];
    const float* maskp    = (const float*)d_in[3];
    const float* msg_w1   = (const float*)d_in[4];
    const float* msg_b1   = (const float*)d_in[5];
    const float* msg_w2   = (const float*)d_in[6];
    const float* msg_b2   = (const float*)d_in[7];
    const float* msg_w3   = (const float*)d_in[8];
    const float* msg_b3   = (const float*)d_in[9];
    const float* n1_scale = (const float*)d_in[10];
    const float* n1_bias  = (const float*)d_in[11];
    const float* dense_w1 = (const float*)d_in[12];
    const float* dense_b1 = (const float*)d_in[13];
    const float* dense_w2 = (const float*)d_in[14];
    const float* dense_b2 = (const float*)d_in[15];
    const float* n2_scale = (const float*)d_in[16];
    const float* n2_bias  = (const float*)d_in[17];
    _Float16* wsf = (_Float16*)d_ws;          // needs 3*448*1024 = 1,376,256 B

    const int n_nodes = in_sizes[0] / CCH;    // 8192

    // 1) repack all fp32 weights into f16 WMMA B-fragment order (L2 resident)
    repack_kernel<<<3*FRAGS_PER_LAYER, 32, 0, stream>>>(
        msg_w1, msg_w2, msg_w3, dense_w1, dense_w2, wsf);

    // 2) fused 3-layer decoder, one workgroup per node
    const size_t smem =
        (size_t)(48*LDA + 48*LDM + 16*LDH + 16*LDD) * sizeof(_Float16)
        + (128*3 + 256 + 256) * sizeof(float);   // 87,552 B
    decoder_kernel<<<n_nodes, 256, smem, stream>>>(
        nodef, edgef, nidx, maskp, msg_b1, msg_b2, msg_b3,
        n1_scale, n1_bias, dense_b1, dense_b2, n2_scale, n2_bias,
        wsf, (float*)d_out);
}
